// Pooling_24850680775446
// MI455X (gfx1250) — 8-bit matrix operand order repaired and checked by reading, compiled, not run on hardware
//
#include <hip/hip_runtime.h>

#define N_NODES 4096
#define N_EDGES 131072
#define D_FEAT  512
#define K_SEL   2048

typedef __attribute__((ext_vector_type(8)))  float v8f;
typedef __attribute__((ext_vector_type(16))) int   v16i;
typedef __attribute__((ext_vector_type(8)))  int   v8i;
typedef __attribute__((ext_vector_type(4)))  int   v4i;
typedef __attribute__((ext_vector_type(2)))  int   v2i;
typedef __attribute__((ext_vector_type(4)))  float f4;
typedef __attribute__((ext_vector_type(4)))  unsigned int u4;

__global__ void k_zero(u4* __restrict__ p, long n) {
    long i = blockIdx.x * (long)blockDim.x + threadIdx.x;
    long stride = gridDim.x * (long)blockDim.x;
    u4 z = (u4)0u;
    for (; i < n; i += stride) p[i] = z;
}

__global__ void k_edges(const int* __restrict__ ei, const float* __restrict__ att,
                        unsigned char* __restrict__ A, unsigned char* __restrict__ At,
                        float* __restrict__ att_sum, float* __restrict__ degree) {
    int e = blockIdx.x * blockDim.x + threadIdx.x;
    if (e >= N_EDGES) return;
    int s = ei[e];
    int d = ei[N_EDGES + e];
    atomicAdd(&att_sum[s], att[e]);
    atomicAdd(&degree[s], 1.0f);
    A [(size_t)s * N_NODES + d] = 0x38;
    At[(size_t)d * N_NODES + s] = 0x38;
}

__global__ void k_scores(const float* __restrict__ att_sum, const float* __restrict__ degree,
                         const int* __restrict__ dir, float* __restrict__ scores) {
    int i = blockIdx.x * blockDim.x + threadIdx.x;
    if (i >= N_NODES) return;
    float deg  = degree[i];
    float mean = (deg > 0.0f) ? (att_sum[i] / fmaxf(deg, 1.0f)) : 0.0f;
    scores[i]  = (float)dir[0] * mean * deg;
}

__global__ void k_topk(const float* __restrict__ scores, const int* __restrict__ batch,
                       int* __restrict__ perm, float* __restrict__ out_batch,
                       float* __restrict__ out_perm) {
    __shared__ float s[N_NODES];
    for (int t = threadIdx.x; t < N_NODES; t += blockDim.x) s[t] = scores[t];
    __syncthreads();
    int i = blockIdx.x * blockDim.x + threadIdx.x;
    float si = s[i];
    int rank = 0;
    for (int j = 0; j < N_NODES; ++j) {
        float sj = s[j];
        rank += (int)((sj > si) || ((sj == si) && (j < i)));
    }
    if (rank < K_SEL) {
        perm[rank]      = i;
        out_perm[rank]  = (float)i;
        out_batch[rank] = (float)batch[i];
    }
}

__global__ void k_gather_x(const float* __restrict__ x, const int* __restrict__ perm,
                           float* __restrict__ out_x) {
    int idx  = blockIdx.x * blockDim.x + threadIdx.x;
    int row  = idx >> 7;
    int col4 = idx & 127;
    const f4* src = reinterpret_cast<const f4*>(x + (size_t)perm[row] * D_FEAT);
    reinterpret_cast<f4*>(out_x + (size_t)row * D_FEAT)[col4] = src[col4];
}

__device__ __forceinline__ v8i load_fragA64(const unsigned char* p, int khalf) {
    v2i c0 = *reinterpret_cast<const v2i*>(p +      8 * khalf);
    v2i c1 = *reinterpret_cast<const v2i*>(p + 16 + 8 * khalf);
    v2i c2 = *reinterpret_cast<const v2i*>(p + 32 + 8 * khalf);
    v2i c3 = *reinterpret_cast<const v2i*>(p + 48 + 8 * khalf);
    v4i lo = __builtin_shufflevector(c0, c1, 0, 1, 2, 3);
    v4i hi = __builtin_shufflevector(c2, c3, 0, 1, 2, 3);
    return __builtin_shufflevector(lo, hi, 0, 1, 2, 3, 4, 5, 6, 7);
}
__device__ __forceinline__ v8i load_fragB64(const unsigned char* p, int khalf) {
    return load_fragA64(p, khalf);
}
__device__ __forceinline__ v16i load_fragA128(const unsigned char* p, int khalf) {
    v8i lo = load_fragA64(p,      khalf);
    v8i hi = load_fragA64(p + 64, khalf);
    return __builtin_shufflevector(lo, hi, 0, 1, 2, 3, 4, 5, 6, 7,
                                           8, 9, 10, 11, 12, 13, 14, 15);
}
__device__ __forceinline__ v16i load_fragB128(const unsigned char* p, int khalf) {
    v8i lo = load_fragB64(p,      khalf);
    v8i hi = load_fragB64(p + 64, khalf);
    return __builtin_shufflevector(lo, hi, 0, 1, 2, 3, 4, 5, 6, 7,
                                           8, 9, 10, 11, 12, 13, 14, 15);
}

__global__ void k_gemm2hop(const unsigned char* __restrict__ A,
                           const unsigned char* __restrict__ At,
                           const int* __restrict__ perm,
                           float* __restrict__ out_adj) {
    int w     = blockIdx.x * 8 + (threadIdx.x >> 5);
    int lane  = threadIdx.x & 31;
    int wy    = w >> 6;
    int wx    = w & 63;
    int i0    = wy * 32;
    int j0    = wx * 32;
    int r     = lane & 15;
    int khalf = lane >> 4;

    const unsigned char* rowA0 = A  + (size_t)perm[i0 + r]      * N_NODES;
    const unsigned char* rowA1 = A  + (size_t)perm[i0 + 16 + r] * N_NODES;
    const unsigned char* rowB0 = At + (size_t)perm[j0 + r]      * N_NODES;
    const unsigned char* rowB1 = At + (size_t)perm[j0 + 16 + r] * N_NODES;

    v8f c00 = {}, c01 = {}, c10 = {}, c11 = {};
    for (int k0 = 0; k0 < N_NODES; k0 += 128) {
        v16i a0 = load_fragA128(rowA0 + k0, khalf);
        v16i a1 = load_fragA128(rowA1 + k0, khalf);
        v16i b0 = load_fragB128(rowB0 + k0, khalf);
        v16i b1 = load_fragB128(rowB1 + k0, khalf);
        c00 = __builtin_amdgcn_wmma_f32_16x16x128_fp8_fp8(a0, b0, (short)0, c00, false, false);
        c01 = __builtin_amdgcn_wmma_f32_16x16x128_fp8_fp8(a0, b1, (short)0, c01, false, false);
        c10 = __builtin_amdgcn_wmma_f32_16x16x128_fp8_fp8(a1, b0, (short)0, c10, false, false);
        c11 = __builtin_amdgcn_wmma_f32_16x16x128_fp8_fp8(a1, b1, (short)0, c11, false, false);
    }

    int n  = lane & 15;
    int mb = 8 * khalf;
#pragma unroll
    for (int t = 0; t < 8; ++t) {
        int m = mb + t;
        out_adj[(size_t)(i0 + m)      * K_SEL + (j0 + n)]      = (c00[t] > 0.0f) ? 1.0f : 0.0f;
        out_adj[(size_t)(i0 + m)      * K_SEL + (j0 + 16 + n)] = (c01[t] > 0.0f) ? 1.0f : 0.0f;
        out_adj[(size_t)(i0 + 16 + m) * K_SEL + (j0 + n)]      = (c10[t] > 0.0f) ? 1.0f : 0.0f;
        out_adj[(size_t)(i0 + 16 + m) * K_SEL + (j0 + 16 + n)] = (c11[t] > 0.0f) ? 1.0f : 0.0f;
    }
}

extern "C" void kernel_launch(void* const* d_in, const int* in_sizes, int n_in,
                              void* d_out, int out_size, void* d_ws, size_t ws_size,
                              hipStream_t stream) {
    (void)in_sizes; (void)n_in; (void)out_size; (void)ws_size;

    const float* x     = (const float*)d_in[0];
    const int*   ei    = (const int*)d_in[1];
    const float* att   = (const float*)d_in[2];
    const int*   batch = (const int*)d_in[3];
    const int*   dir   = (const int*)d_in[4];

    char* ws = (char*)d_ws;
    unsigned char* A       = (unsigned char*)(ws);
    unsigned char* At      = (unsigned char*)(ws + 16777216);
    float*         att_sum = (float*)(ws + 33554432);
    float*         degree  = (float*)(ws + 33570816);
    float*         scores  = (float*)(ws + 33587200);
    int*           perm    = (int*)  (ws + 33603584);

    float* out       = (float*)d_out;
    float* out_x     = out;
    float* out_adj   = out + (size_t)K_SEL * D_FEAT;
    float* out_batch = out_adj + (size_t)K_SEL * K_SEL;
    float* out_perm  = out_batch + K_SEL;

    k_zero<<<1024, 256, 0, stream>>>((u4*)ws, 33587200L / 16L);
    k_edges<<<N_EDGES / 256, 256, 0, stream>>>(ei, att, A, At, att_sum, degree);
    k_scores<<<N_NODES / 256, 256, 0, stream>>>(att_sum, degree, dir, scores);
    k_topk<<<N_NODES / 256, 256, 0, stream>>>(scores, batch, perm, out_batch, out_perm);
    k_gather_x<<<(K_SEL * D_FEAT / 4) / 256, 256, 0, stream>>>(x, perm, out_x);
    k_gemm2hop<<<(64 * 64) / 8, 256, 0, stream>>>(A, At, perm, out_adj);
}
